// GLALayer_62302795596646
// MI455X (gfx1250) — matrix operand order repaired and checked by reading, not compiled, not run on hardware
//
#include <hip/hip_runtime.h>
#include <hip/hip_bf16.h>

// ---------------------------------------------------------------------------
// GLA layer for MI455X (gfx1250, wave32, WMMA).
//   B=4, T=2048, d=1024, H=16, dk=dv=64
// Pipeline:
//   1) convert_pack : fp32 -> bf16; weights pre-TRANSPOSED to [N][K] so both
//                     A and B WMMA fragments are contiguous-dword LDS loads
//   2) gemm_wmma    : qkva = act(x @ Wcat) (sigmoid+bias on gate cols), bf16
//   3) gla_scan     : per-(b,h) recurrence, state in VGPRs, bf16 o out
//   4) gemm_wmma    : out = o @ Wo, fp32 out
// GEMM: 128x256 block tile, 8 waves x (64x64), BK=32, double-buffered LDS
//       staged with GLOBAL_LOAD_ASYNC_TO_LDS_B128 when available.
// ---------------------------------------------------------------------------

typedef __attribute__((ext_vector_type(16))) __bf16 v16bf;
typedef __attribute__((ext_vector_type(8)))  float  v8f;
typedef int v4i __attribute__((vector_size(16)));   // matches async builtin

#define GLA_B   4
#define GLA_T   2048
#define GLA_D   1024
#define GLA_H   16
#define GLA_DK  64
#define GLA_M   (GLA_B * GLA_T)        // 8192 rows
#define GLA_N4  (4 * GLA_D)            // 4096 fused output cols

#define BM 128
#define BN 256
#define BK 32
#define LDA 40                          // padded LDS row stride (bf16 elems)

#if __has_builtin(__builtin_amdgcn_global_load_async_to_lds_b128) && \
    __has_builtin(__builtin_amdgcn_s_wait_asynccnt)
#define GLA_USE_ASYNC 1
#endif

#define AS_G __attribute__((address_space(1)))
#define AS_L __attribute__((address_space(3)))

// ---------------------------------------------------------------------------
// Kernel 1: convert + pack; weights stored transposed [N][K]
// ---------------------------------------------------------------------------
__global__ void gla_convert_pack(const float* __restrict__ x,
                                 const float* __restrict__ Wq,
                                 const float* __restrict__ Wk,
                                 const float* __restrict__ Wv,
                                 const float* __restrict__ Wg,
                                 const float* __restrict__ Wo,
                                 __bf16* __restrict__ xb,
                                 __bf16* __restrict__ wcatT,   // [4096][1024]
                                 __bf16* __restrict__ wobT) {  // [1024][1024]
    const size_t NX = (size_t)GLA_M * GLA_D;
    const size_t NW = (size_t)GLA_N4 * GLA_D;
    const size_t NO = (size_t)GLA_D * GLA_D;
    size_t i = ((size_t)blockIdx.x * blockDim.x + threadIdx.x) * 4;
    if (i >= NX + NW + NO) return;

    union { __bf16 h[4]; uint2 u; } pk;
    if (i < NX) {
        float4 f = *(const float4*)(x + i);
        pk.h[0] = (__bf16)f.x; pk.h[1] = (__bf16)f.y;
        pk.h[2] = (__bf16)f.z; pk.h[3] = (__bf16)f.w;
        *(uint2*)(xb + i) = pk.u;
    } else if (i < NX + NW) {
        size_t j = i - NX;                 // linear index into wcatT[n][k]
        int n = (int)(j >> 10), k = (int)(j & 1023);
        int s = n >> 10, c = n & 1023;
        const float* w = (s == 0) ? Wq : (s == 1) ? Wk : (s == 2) ? Wv : Wg;
#pragma unroll
        for (int u = 0; u < 4; ++u) pk.h[u] = (__bf16)w[(size_t)(k + u) * GLA_D + c];
        *(uint2*)(wcatT + j) = pk.u;
    } else {
        size_t j = i - NX - NW;
        int n = (int)(j >> 10), k = (int)(j & 1023);
#pragma unroll
        for (int u = 0; u < 4; ++u) pk.h[u] = (__bf16)Wo[(size_t)(k + u) * GLA_D + n];
        *(uint2*)(wobT + j) = pk.u;
    }
}

// ---------------------------------------------------------------------------
// Kernel 2: bf16 WMMA GEMM, C[M,N] = act(A[M,K] @ BT[N,K]^T)
//   mode 0: bf16 store; cols >= 3072 get sigmoid(x + bias[col-3072])
//   mode 1: fp32 store, no activation
// ---------------------------------------------------------------------------
__global__ __launch_bounds__(256)
void gla_gemm_wmma(const __bf16* __restrict__ A,     // [M][K]
                   const __bf16* __restrict__ BT,    // [N][K]
                   void* __restrict__ C,
                   const float* __restrict__ bias,
                   int M, int N, int K, int mode) {
    __shared__ union {
        struct {
            __bf16 a[2][BM][LDA];     // 20 KB
            __bf16 b[2][BN][LDA];     // 40 KB
        } t;
        float ep[8][32 * 64];          // 64 KB epilogue slices (8 waves)
    } sm;

    const int tid = threadIdx.x;
    const int m0  = blockIdx.y * BM;
    const int n0  = blockIdx.x * BN;

    const int l   = tid & 31;
    const int w   = tid >> 5;
    const int wm  = (w >> 2) * 64;     // wave row base in block tile
    const int wn  = (w & 3) * 64;      // wave col base in block tile
    const int r15 = l & 15;
    const int hi  = l >> 4;

    // staging chunk coords: both tiles are <rows> x 32 bf16, 8-elem chunks
    const int r0 = tid >> 2;           // 0..63
    const int c0 = (tid & 3) * 8;      // 0,8,16,24
    const __bf16* gA0 = A  + (size_t)(m0 + r0) * K + c0;
    const __bf16* gA1 = gA0 + (size_t)64 * K;
    const __bf16* gB0 = BT + (size_t)(n0 + r0) * K + c0;
    const __bf16* gB1 = gB0 + (size_t)64  * K;
    const __bf16* gB2 = gB0 + (size_t)128 * K;
    const __bf16* gB3 = gB0 + (size_t)192 * K;

    v8f acc[4][4];
#pragma unroll
    for (int mt = 0; mt < 4; ++mt)
#pragma unroll
        for (int nt = 0; nt < 4; ++nt)
#pragma unroll
            for (int e = 0; e < 8; ++e) acc[mt][nt][e] = 0.0f;

    const int NT = K / BK;             // 32

    // ---- fragment load + 16 WMMAs on buffer `buf` -----------------------
    auto compute = [&](int buf) {
        union { v16bf v; unsigned int u[8]; } af[4], bfr[4];
#pragma unroll
        for (int mt = 0; mt < 4; ++mt) {
            const int rb = wm + mt * 16 + r15;
            const int kh = hi * 8;
#pragma unroll
            for (int i = 0; i < 8; ++i) {
                const int kk = kh + ((i < 4) ? 2 * i : 16 + 2 * (i - 4));
                af[mt].u[i] = *(const unsigned int*)&sm.t.a[buf][rb][kk];
            }
        }
#pragma unroll
        for (int nt = 0; nt < 4; ++nt) {
            const int cc = wn + nt * 16 + r15;
            const int kb = hi * 8;
#pragma unroll
            for (int i = 0; i < 8; ++i)
                bfr[nt].u[i] = *(const unsigned int*)&sm.t.b[buf][cc][kb + ((i < 4) ? 2 * i : 16 + 2 * (i - 4))];
        }
#pragma unroll
        for (int mt = 0; mt < 4; ++mt)
#pragma unroll
            for (int nt = 0; nt < 4; ++nt)
                acc[mt][nt] = __builtin_amdgcn_wmma_f32_16x16x32_bf16(
                    false, af[mt].v, false, bfr[nt].v,
                    (short)0, acc[mt][nt], false, false);
    };

#ifdef GLA_USE_ASYNC
    // ---- double-buffered async global->LDS staging ----------------------
    auto stage_async = [&](int buf, int k0) {
        __builtin_amdgcn_global_load_async_to_lds_b128(
            (AS_G v4i*)(gA0 + k0), (AS_L v4i*)&sm.t.a[buf][r0][c0], 0, 0);
        __builtin_amdgcn_global_load_async_to_lds_b128(
            (AS_G v4i*)(gA1 + k0), (AS_L v4i*)&sm.t.a[buf][r0 + 64][c0], 0, 0);
        __builtin_amdgcn_global_load_async_to_lds_b128(
            (AS_G v4i*)(gB0 + k0), (AS_L v4i*)&sm.t.b[buf][r0][c0], 0, 0);
        __builtin_amdgcn_global_load_async_to_lds_b128(
            (AS_G v4i*)(gB1 + k0), (AS_L v4i*)&sm.t.b[buf][r0 + 64][c0], 0, 0);
        __builtin_amdgcn_global_load_async_to_lds_b128(
            (AS_G v4i*)(gB2 + k0), (AS_L v4i*)&sm.t.b[buf][r0 + 128][c0], 0, 0);
        __builtin_amdgcn_global_load_async_to_lds_b128(
            (AS_G v4i*)(gB3 + k0), (AS_L v4i*)&sm.t.b[buf][r0 + 192][c0], 0, 0);
    };

    stage_async(0, 0);
    for (int i = 0; i < NT - 1; ++i) {
        stage_async((i + 1) & 1, (i + 1) * BK);       // 6 outstanding on top
        __builtin_amdgcn_s_wait_asynccnt(6);          // tile i landed (per-wave)
        __syncthreads();                              // ... for all waves
        compute(i & 1);
        __syncthreads();                              // safe to overwrite buf
    }
    __builtin_amdgcn_s_wait_asynccnt(0);
    __syncthreads();
    compute((NT - 1) & 1);
#else
    // ---- register-pipelined sync staging fallback -----------------------
    uint4 ra0, ra1, rb0, rb1, rb2, rb3;
    auto g2r = [&](int k0) {
        ra0 = *(const uint4*)(gA0 + k0); ra1 = *(const uint4*)(gA1 + k0);
        rb0 = *(const uint4*)(gB0 + k0); rb1 = *(const uint4*)(gB1 + k0);
        rb2 = *(const uint4*)(gB2 + k0); rb3 = *(const uint4*)(gB3 + k0);
        __builtin_prefetch((const void*)(gA0 + k0 + BK), 0, 1);
        __builtin_prefetch((const void*)(gB0 + k0 + BK), 0, 1);
    };
    auto r2l = [&](int buf) {
        *(uint4*)&sm.t.a[buf][r0][c0]       = ra0;
        *(uint4*)&sm.t.a[buf][r0 + 64][c0]  = ra1;
        *(uint4*)&sm.t.b[buf][r0][c0]       = rb0;
        *(uint4*)&sm.t.b[buf][r0 + 64][c0]  = rb1;
        *(uint4*)&sm.t.b[buf][r0 + 128][c0] = rb2;
        *(uint4*)&sm.t.b[buf][r0 + 192][c0] = rb3;
    };
    g2r(0); r2l(0); __syncthreads();
    for (int i = 0; i < NT; ++i) {
        if (i + 1 < NT) g2r((i + 1) * BK);
        compute(i & 1);
        __syncthreads();
        if (i + 1 < NT) { r2l((i + 1) & 1); __syncthreads(); }
    }
#endif

    // ---- epilogue: restage through LDS, coalesced b128 stores -----------
    __bf16* cbf = (__bf16*)C;
    float*  cf  = (float*)C;
    float*  slice = sm.ep[w];                    // 32x64 f32 per wave
    const int gnb = n0 + wn;
    __syncthreads();                             // done with tile buffers

#pragma unroll
    for (int pass = 0; pass < 2; ++pass) {       // two 32-row halves
#pragma unroll
        for (int mt2 = 0; mt2 < 2; ++mt2) {
            const int mt = pass * 2 + mt2;
#pragma unroll
            for (int nt = 0; nt < 4; ++nt)
#pragma unroll
                for (int v = 0; v < 8; ++v)
                    slice[(mt2 * 16 + hi * 8 + v) * 64 + nt * 16 + r15] =
                        acc[mt][nt][v];
        }
        __syncthreads();                         // LDS writes visible
        const int gmb = m0 + wm + pass * 32;
        if (mode == 0) {
#pragma unroll
            for (int j = 0; j < 8; ++j) {
                const int e  = (l + j * 32) * 8;
                const int lr = e >> 6, lc = e & 63;
                float4 x0 = *(const float4*)&slice[e];
                float4 x1 = *(const float4*)&slice[e + 4];
                float vals[8] = {x0.x, x0.y, x0.z, x0.w, x1.x, x1.y, x1.z, x1.w};
                const int gn0 = gnb + lc;
                if (gn0 >= 3072) {               // gate block: sigmoid(x + b)
#pragma unroll
                    for (int u = 0; u < 8; ++u) {
                        float t = vals[u] + bias[gn0 - 3072 + u];
                        vals[u] = 1.0f / (1.0f + __expf(-t));
                    }
                }
                union { __bf16 h[8]; uint4 u4; } pk;
#pragma unroll
                for (int u = 0; u < 8; ++u) pk.h[u] = (__bf16)vals[u];
                *(uint4*)(cbf + (size_t)(gmb + lr) * N + gn0) = pk.u4;
            }
        } else {
#pragma unroll
            for (int j = 0; j < 16; ++j) {
                const int e  = (l + j * 32) * 4;
                const int lr = e >> 6, lc = e & 63;
                *(float4*)(cf + (size_t)(gmb + lr) * N + gnb + lc) =
                    *(const float4*)&slice[e];
            }
        }
        __syncthreads();                         // before next pass overwrite
    }
}

// ---------------------------------------------------------------------------
// Kernel 3: gated scan. One block (64 threads) per (b,h); thread = dv column.
// S[i][c] = a_i * S[i][c] + k_i * v_c ; o_c = sum_i q_i * S[i][c]
// k/a/q vectors are block-uniform -> scalar-load broadcast; state in VGPRs.
// ---------------------------------------------------------------------------
__global__ __launch_bounds__(64)
void gla_scan(const __bf16* __restrict__ qkva, __bf16* __restrict__ obuf) {
    const int bh = blockIdx.x;
    const int b  = bh >> 4;
    const int h  = bh & 15;
    const int c  = threadIdx.x;

    float S[GLA_DK];
#pragma unroll
    for (int i = 0; i < GLA_DK; ++i) S[i] = 0.0f;

    const __bf16* base = qkva + (size_t)b * GLA_T * GLA_N4 + h * GLA_DK;
    __bf16* ob = obuf + (size_t)b * GLA_T * GLA_D + h * GLA_DK + c;

    for (int t = 0; t < GLA_T; ++t) {
        const __bf16* row = base + (size_t)t * GLA_N4;
        const float vc = (float)row[2 * GLA_D + c];
        const unsigned int* qrow = (const unsigned int*)(row);
        const unsigned int* krow = (const unsigned int*)(row + GLA_D);
        const unsigned int* arow = (const unsigned int*)(row + 3 * GLA_D);

        float out = 0.0f;
#pragma unroll
        for (int p = 0; p < GLA_DK / 2; ++p) {
            const unsigned int kp = krow[p];           // uniform -> s_load
            const unsigned int ap = arow[p];
            const unsigned int qp = qrow[p];
            const float k0 = __uint_as_float(kp << 16);
            const float k1 = __uint_as_float(kp & 0xffff0000u);
            const float a0 = __uint_as_float(ap << 16);
            const float a1 = __uint_as_float(ap & 0xffff0000u);
            const float q0 = __uint_as_float(qp << 16);
            const float q1 = __uint_as_float(qp & 0xffff0000u);
            S[2 * p]     = fmaf(a0, S[2 * p],     k0 * vc);
            S[2 * p + 1] = fmaf(a1, S[2 * p + 1], k1 * vc);
            out = fmaf(q0, S[2 * p], out);
            out = fmaf(q1, S[2 * p + 1], out);
        }
        ob[(size_t)t * GLA_D] = (__bf16)out;
    }
}

// ---------------------------------------------------------------------------
// Host launcher
// ---------------------------------------------------------------------------
extern "C" void kernel_launch(void* const* d_in, const int* in_sizes, int n_in,
                              void* d_out, int out_size, void* d_ws, size_t ws_size,
                              hipStream_t stream) {
    (void)in_sizes; (void)n_in; (void)out_size; (void)ws_size;
    const float* x  = (const float*)d_in[0];
    const float* Wq = (const float*)d_in[1];
    const float* Wk = (const float*)d_in[2];
    const float* Wv = (const float*)d_in[3];
    const float* Wg = (const float*)d_in[4];
    const float* bg = (const float*)d_in[5];
    const float* Wo = (const float*)d_in[6];

    char* p = (char*)d_ws;
    __bf16* xb    = (__bf16*)p; p += (size_t)GLA_M * GLA_D * 2;     // 16 MB
    __bf16* wcatT = (__bf16*)p; p += (size_t)GLA_N4 * GLA_D * 2;    //  8 MB
    __bf16* wobT  = (__bf16*)p; p += (size_t)GLA_D * GLA_D * 2;     //  2 MB
    __bf16* qkva  = (__bf16*)p; p += (size_t)GLA_M * GLA_N4 * 2;    // 64 MB
    __bf16* obuf  = (__bf16*)p; p += (size_t)GLA_M * GLA_D * 2;     // 16 MB

    {
        const size_t total = (size_t)GLA_M * GLA_D + (size_t)GLA_N4 * GLA_D
                           + (size_t)GLA_D * GLA_D;
        const int grid = (int)((total / 4 + 255) / 256);
        gla_convert_pack<<<grid, 256, 0, stream>>>(x, Wq, Wk, Wv, Wg, Wo,
                                                   xb, wcatT, wobT);
    }
    gla_gemm_wmma<<<dim3(GLA_N4 / BN, GLA_M / BM), 256, 0, stream>>>(
        xb, wcatT, (void*)qkva, bg, GLA_M, GLA_N4, GLA_D, 0);
    gla_scan<<<GLA_B * GLA_H, 64, 0, stream>>>(qkva, obuf);
    gla_gemm_wmma<<<dim3(GLA_D / BN, GLA_M / BM), 256, 0, stream>>>(
        obuf, wobT, d_out, nullptr, GLA_M, GLA_D, GLA_D, 1);
}
